// MultiHeadSelfAttention_3418793967657
// MI455X (gfx1250) — matrix operand order repaired and checked by reading, not compiled, not run on hardware
//
#include <hip/hip_runtime.h>
#include <hip/hip_bf16.h>

#define BATCH 2
#define SEQ   4096
#define DIN   256
#define DKT   256   // d_model
#define NHEAD 8
#define HD    32    // head dim
#define KT    64    // key tile per block iteration

typedef __attribute__((ext_vector_type(16))) __bf16         v16bf;
typedef __attribute__((ext_vector_type(8)))  float          v8f;
typedef __attribute__((ext_vector_type(16))) unsigned short us16;
typedef __attribute__((ext_vector_type(8)))  unsigned short us8;
typedef __attribute__((ext_vector_type(4)))  unsigned short us4;
typedef __attribute__((ext_vector_type(4)))  float          f4;

// fp32 -> bf16 round-to-nearest-even
__device__ __forceinline__ unsigned short f2bf(float f) {
    unsigned int u = __builtin_bit_cast(unsigned int, f);
    u += 0x7FFFu + ((u >> 16) & 1u);
    return (unsigned short)(u >> 16);
}

__device__ __forceinline__ v8f wmma_bf16(us16 a, us16 b, v8f c) {
    return __builtin_amdgcn_wmma_f32_16x16x32_bf16(
        false, __builtin_bit_cast(v16bf, a),
        false, __builtin_bit_cast(v16bf, b),
        (short)0, c, false, false);
}

// Issue one per-lane async 16B global->LDS copy (ASYNCcnt-tracked).
__device__ __forceinline__ void async_b128(const void* gptr, const void* ldsptr) {
    unsigned int loff = (unsigned int)(unsigned long long)ldsptr;  // low 32b = LDS byte offset
    asm volatile("global_load_async_to_lds_b128 %0, %1, off"
                 :: "v"(loff), "v"((unsigned long long)gptr) : "memory");
}

// ---------------------------------------------------------------------------
// Kernel 0a: x (fp32) -> xb (bf16), same layout.  4 elems/thread.
// ---------------------------------------------------------------------------
__global__ __launch_bounds__(256)
void cvt_x_kernel(const float* __restrict__ x, unsigned short* __restrict__ xb) {
    const size_t i = ((size_t)blockIdx.x * 256 + threadIdx.x) * 4;
    f4 v = *(const f4*)(x + i);
    us4 o;
#pragma unroll
    for (int j = 0; j < 4; ++j) o[j] = f2bf(v[j]);
    *(us4*)(xb + i) = o;
}

// ---------------------------------------------------------------------------
// Kernel 0b: W (fp32, [i][c]) -> Wt (bf16, transposed [c][i]) for Q,K,V.
// gridDim.y = which matrix.
// ---------------------------------------------------------------------------
__global__ __launch_bounds__(256)
void cvt_w_kernel(const float* __restrict__ Wq,
                  const float* __restrict__ Wk,
                  const float* __restrict__ Wv,
                  unsigned short* __restrict__ wt) {
    const int which = blockIdx.y;
    const float* W = (which == 0) ? Wq : (which == 1) ? Wk : Wv;
    const int idx = blockIdx.x * 256 + threadIdx.x;       // [0, 65536)
    const int c = idx & 255;                              // coalesced read
    const int i = idx >> 8;
    wt[(size_t)which * DIN * DKT + (size_t)c * DIN + i] = f2bf(W[(size_t)i * DKT + c]);
}

// ---------------------------------------------------------------------------
// Kernel 1: QKV projection (xb @ W -> bf16), one 16x16 tile per wave.
// All operands pre-converted bf16; B rows contiguous via transposed Wt.
// which = blockIdx.y (scalar).  V stored transposed [B,NH,HD,N].
// ---------------------------------------------------------------------------
__global__ __launch_bounds__(128)
void qkv_proj_kernel(const unsigned short* __restrict__ xb,
                     const unsigned short* __restrict__ wt,
                     unsigned short* __restrict__ qws,
                     unsigned short* __restrict__ kws,
                     unsigned short* __restrict__ vtws) {
    const int lane = threadIdx.x & 31;
    const int wid  = blockIdx.x * 4 + (threadIdx.x >> 5);   // [0, 8192)
    const int which = blockIdx.y;                           // 0=Q 1=K 2=V (scalar)
    const int ntile = wid >> 4;                             // 512 row tiles
    const int ctile = wid & 15;                             // 16 col tiles
    const int lm = lane & 15;
    const int hs = lane >> 4;

    const unsigned short* Wt = wt + (size_t)which * DIN * DKT;
    const int r = ntile * 16 + lm;                          // A row (token)
    const int c = ctile * 16 + lm;                          // B col
    const unsigned short* xr = xb + (size_t)r * DIN;
    const unsigned short* wr = Wt + (size_t)c * DIN;        // B column, contiguous

    v8f acc = {};
    for (int ib = 0; ib < DIN; ib += 32) {
        // A: x tile in WMMA A layout (two 16B chunks)
        us8 xa0 = *(const us8*)(xr + ib + 16 * hs);
        us8 xa1 = *(const us8*)(xr + ib + 16 * hs + 8);
        us16 ua;
#pragma unroll
        for (int j = 0; j < 8; ++j) { ua[j] = xa0[j]; ua[8 + j] = xa1[j]; }

        // B: 16 contiguous bf16 (k = ib + 16*hs .. +15)
        us16 ub = *(const us16*)(wr + ib + 16 * hs);

        acc = wmma_bf16(ua, ub, acc);
    }

    const int h = c >> 5, d = c & 31;
#pragma unroll
    for (int v = 0; v < 8; ++v) {
        const int r2 = ntile * 16 + v + 8 * hs;
        const int b  = r2 >> 12;
        const int nn = r2 & (SEQ - 1);
        const unsigned short bv = f2bf(acc[v]);
        if (which == 0)
            qws[((size_t)(b * NHEAD + h) * SEQ + nn) * HD + d] = bv;
        else if (which == 1)
            kws[((size_t)(b * NHEAD + h) * SEQ + nn) * HD + d] = bv;
        else
            vtws[((size_t)(b * NHEAD + h) * HD + d) * SEQ + nn] = bv;
    }
}

// ---------------------------------------------------------------------------
// Kernel 2: flash attention, block = 8 waves = 128 queries of one (b,h).
// K/V tiles (64 keys) staged global->LDS with async copies, double-buffered;
// per wave: 4 WMMAs for S, online softmax (shfl_xor over 16-lane halves),
// P transposed through LDS (C->A layout), 4 WMMAs for O += P.V.
// ---------------------------------------------------------------------------
__global__ __launch_bounds__(256)
void flash_attn_kernel(const unsigned short* __restrict__ qws,
                       const unsigned short* __restrict__ kws,
                       const unsigned short* __restrict__ vtws,
                       float* __restrict__ out) {
    __shared__ unsigned short ldsK[2][KT][HD];   // 8 KB : keys x dk
    __shared__ unsigned short ldsV[2][HD][KT];   // 8 KB : dv x keys
    __shared__ unsigned short Pl[8][16][KT];     // 16 KB: per-wave P tile

    const int tid  = threadIdx.x;
    const int wave = tid >> 5, lane = tid & 31;
    const int lm = lane & 15, hs = lane >> 4;
    const int bx   = blockIdx.x;
    const int qblk = bx & 31;
    const int h    = (bx >> 5) & 7;
    const int b    = bx >> 8;
    const int q0   = qblk * 128 + wave * 16;

    const unsigned short* Qbh = qws  + (size_t)(b * NHEAD + h) * SEQ * HD;
    const unsigned short* Kbh = kws  + (size_t)(b * NHEAD + h) * SEQ * HD;
    const unsigned short* Vbh = vtws + (size_t)(b * NHEAD + h) * HD * SEQ;

    // cooperative async stage of one 64-key K/V tile (2 loads per lane)
    auto stage = [&](int buf, int kb) {
        {   // K tile: 64 rows x 64B -> 256 x 16B chunks
            const int r = tid >> 2, ch = (tid & 3) * 8;
            async_b128(Kbh + (size_t)(kb + r) * HD + ch, &ldsK[buf][r][ch]);
        }
        {   // V tile: 32 rows x 128B -> 256 x 16B chunks
            const int d = tid >> 3, ch = (tid & 7) * 8;
            async_b128(Vbh + (size_t)d * SEQ + kb + ch, &ldsV[buf][d][ch]);
        }
    };

    // Q tile once, WMMA A layout (dk=32 == one WMMA K)
    const unsigned short* qp = Qbh + (size_t)(q0 + lm) * HD + 16 * hs;
    us8 qa0 = *(const us8*)qp;
    us8 qa1 = *(const us8*)(qp + 8);
    us16 aQ;
#pragma unroll
    for (int j = 0; j < 8; ++j) { aQ[j] = qa0[j]; aQ[8 + j] = qa1[j]; }

    float mst[8], lst[8];
#pragma unroll
    for (int v = 0; v < 8; ++v) { mst[v] = -1e30f; lst[v] = 0.0f; }
    v8f O0 = {}, O1 = {};
    const float scale = 0.17677669529663688f;   // 1/sqrt(32)

    stage(0, 0);
    int buf = 0;
    for (int kb = 0; kb < SEQ; kb += KT) {
        if (kb + KT < SEQ) {
            stage(buf ^ 1, kb + KT);                        // prefetch next tile
            asm volatile("s_wait_asynccnt 2" ::: "memory"); // current tile landed
        } else {
            asm volatile("s_wait_asynccnt 0" ::: "memory");
        }
        __syncthreads();

        // K as B-matrix: 4 groups of 16 keys
        us16 kr0 = *(const us16*)&ldsK[buf][ 0 + lm][16 * hs];
        us16 kr1 = *(const us16*)&ldsK[buf][16 + lm][16 * hs];
        us16 kr2 = *(const us16*)&ldsK[buf][32 + lm][16 * hs];
        us16 kr3 = *(const us16*)&ldsK[buf][48 + lm][16 * hs];
        v8f z = {};
        v8f S0 = wmma_bf16(aQ, kr0, z);
        v8f S1 = wmma_bf16(aQ, kr1, z);
        v8f S2 = wmma_bf16(aQ, kr2, z);
        v8f S3 = wmma_bf16(aQ, kr3, z);

        // V as B-matrix: [kh][nb] = keys 0..31/32..63 x dv 0..15/16..31
        us16 vr00 = *(const us16*)&ldsV[buf][lm     ][ 0 + 16 * hs];
        us16 vr10 = *(const us16*)&ldsV[buf][lm     ][32 + 16 * hs];
        us16 vr01 = *(const us16*)&ldsV[buf][lm + 16][ 0 + 16 * hs];
        us16 vr11 = *(const us16*)&ldsV[buf][lm + 16][32 + 16 * hs];

#pragma unroll
        for (int v = 0; v < 8; ++v) {
            float s0 = S0[v] * scale, s1 = S1[v] * scale;
            float s2 = S2[v] * scale, s3 = S3[v] * scale;
            float t = fmaxf(fmaxf(s0, s1), fmaxf(s2, s3));
            t = fmaxf(t, __shfl_xor(t, 1));
            t = fmaxf(t, __shfl_xor(t, 2));
            t = fmaxf(t, __shfl_xor(t, 4));
            t = fmaxf(t, __shfl_xor(t, 8));
            const float mnew = fmaxf(mst[v], t);
            const float corr = __expf(mst[v] - mnew);
            const float p0 = __expf(s0 - mnew);
            const float p1 = __expf(s1 - mnew);
            const float p2 = __expf(s2 - mnew);
            const float p3 = __expf(s3 - mnew);
            float rs = (p0 + p1) + (p2 + p3);
            rs += __shfl_xor(rs, 1);
            rs += __shfl_xor(rs, 2);
            rs += __shfl_xor(rs, 4);
            rs += __shfl_xor(rs, 8);
            lst[v] = lst[v] * corr + rs;
            mst[v] = mnew;
            O0[v] *= corr;
            O1[v] *= corr;
            const int row = v + 8 * hs;
            Pl[wave][row][lm     ] = f2bf(p0);
            Pl[wave][row][lm + 16] = f2bf(p1);
            Pl[wave][row][lm + 32] = f2bf(p2);
            Pl[wave][row][lm + 48] = f2bf(p3);
        }
        asm volatile("s_wait_dscnt 0" ::: "memory");

        // Reload P in A layout: aP0 = keys 0..31, aP1 = keys 32..63
        const unsigned short* pr = &Pl[wave][lm][16 * hs];
        us8 a0 = *(const us8*)(pr);
        us8 a1 = *(const us8*)(pr + 8);
        us8 a2 = *(const us8*)(pr + 32);
        us8 a3 = *(const us8*)(pr + 40);
        us16 aP0, aP1;
#pragma unroll
        for (int j = 0; j < 8; ++j) {
            aP0[j] = a0[j]; aP0[8 + j] = a1[j];
            aP1[j] = a2[j]; aP1[8 + j] = a3[j];
        }
        O0 = wmma_bf16(aP0, vr00, O0);
        O0 = wmma_bf16(aP1, vr10, O0);
        O1 = wmma_bf16(aP0, vr01, O1);
        O1 = wmma_bf16(aP1, vr11, O1);

        __syncthreads();   // all waves done reading buf before it is restaged
        buf ^= 1;
    }

    // Epilogue: divide by softmax denominator, write fp32
#pragma unroll
    for (int v = 0; v < 8; ++v) {
        const float inv = 1.0f / lst[v];
        const int r = q0 + v + 8 * hs;
        float* op = out + ((size_t)b * SEQ + r) * DKT + h * HD;
        op[lm]      = O0[v] * inv;
        op[lm + 16] = O1[v] * inv;
    }
}

// ---------------------------------------------------------------------------
extern "C" void kernel_launch(void* const* d_in, const int* in_sizes, int n_in,
                              void* d_out, int out_size, void* d_ws, size_t ws_size,
                              hipStream_t stream) {
    const float* x  = (const float*)d_in[0];
    const float* Wq = (const float*)d_in[1];
    const float* Wk = (const float*)d_in[2];
    const float* Wv = (const float*)d_in[3];
    float* out = (float*)d_out;

    const size_t per = (size_t)BATCH * NHEAD * SEQ * HD;   // 2,097,152 bf16 elems
    unsigned short* qws  = (unsigned short*)d_ws;
    unsigned short* kws  = qws + per;
    unsigned short* vtws = kws + per;
    unsigned short* xb   = vtws + per;                     // 2,097,152 elems
    unsigned short* wt   = xb + (size_t)BATCH * SEQ * DIN; // 3*65536 elems

    // Pre-convert activations and (transposed) weights to bf16.
    cvt_x_kernel<<<2048, 256, 0, stream>>>(x, xb);                 // 2M elems / 4
    cvt_w_kernel<<<dim3(256, 3), 256, 0, stream>>>(Wq, Wk, Wv, wt);

    // 8192 tiles per matrix, 4 waves/block; y = which matrix (scalar in-kernel)
    qkv_proj_kernel<<<dim3(2048, 3), 128, 0, stream>>>(xb, wt, qws, kws, vtws);
    // B*NH*(N/128) = 512 blocks, 8 waves each
    flash_attn_kernel<<<512, 256, 0, stream>>>(qws, kws, vtws, out);
}
